// FastLDPCDecoder_4750233829549
// MI455X (gfx1250) — matrix operand order repaired and checked by reading, not compiled, not run on hardware
//
#include <hip/hip_runtime.h>

// ---------------------------------------------------------------------------
// LDPC SPA decoder for MI455X (gfx1250, wave32).
//  - Sparse CSR message passing (check/variable kernels), all data L2-resident
//  - Dense syndrome check bits @ H^T done with v_wmma_f32_16x16x32_f16
//    (bits/H are 0/1 -> exact in f16/f32)
// ---------------------------------------------------------------------------

typedef __attribute__((ext_vector_type(16))) _Float16 v16h;
typedef __attribute__((ext_vector_type(8)))  float    v8f;

#define B_      16
#define N_      1024
#define R_      512
#define MAXDEG  32
#define NITER   20
#define GAMMA_  1.0001f
#define CLIP_   20.0f
#define CONF_   5.0f
#define KT_     (N_ / 32)   // 32 K-tiles of 32
#define JT_     (R_ / 16)   // 32 check tiles of 16

__device__ __forceinline__ float clipf(float x) {
    return fminf(fmaxf(x, -CLIP_), CLIP_);
}
__device__ __forceinline__ float phif(float a) {           // -log(tanh(a/2))
    return -logf(tanhf(0.5f * a));
}

// ---- one-time structure build ---------------------------------------------

__global__ void k_rows(const float* __restrict__ H, int* rowDeg, int* rowCols) {
    int r = blockIdx.x * blockDim.x + threadIdx.x;
    if (r >= R_) return;
    int deg = 0;
    for (int c = 0; c < N_; ++c) {
        if (H[r * N_ + c] != 0.0f) {
            if (deg < MAXDEG) rowCols[r * MAXDEG + deg] = c;
            ++deg;
        }
    }
    rowDeg[r] = (deg < MAXDEG) ? deg : MAXDEG;
}

__global__ void k_cols(const float* __restrict__ H, const int* rowDeg,
                       const int* rowCols, int* colDeg, int* colEdges) {
    int c = blockIdx.x * blockDim.x + threadIdx.x;
    if (c >= N_) return;
    int cd = 0;
    for (int r = 0; r < R_; ++r) {
        if (H[r * N_ + c] != 0.0f) {
            int d = rowDeg[r], s = -1;
            for (int q = 0; q < d; ++q)
                if (rowCols[r * MAXDEG + q] == c) { s = q; break; }
            if (s >= 0 && cd < MAXDEG) colEdges[c * MAXDEG + cd++] = r * MAXDEG + s;
        }
    }
    colDeg[c] = cd;
}

// Pack H^T into WMMA B layout: Bpack[kt][j][lane][e] = H[j*16+(lane&15)][kt*32+K]
// (B is KxN = 32x16 per tile; N = lane&15, K = 8*(lane>>4) + e + (e>=8 ? 8 : 0))
__global__ void k_packB(const float* __restrict__ H, _Float16* Bpack) {
    int t = blockIdx.x * blockDim.x + threadIdx.x;
    if (t >= KT_ * JT_ * 32 * 16) return;
    int e  = t & 15;
    int L  = (t >> 4) & 31;
    int j  = (t >> 9) & 31;
    int kt = t >> 14;
    int rr = j * 16 + (L & 15);
    int nn = kt * 32 + (L >> 4) * 8 + e + ((e >= 8) ? 8 : 0);
    Bpack[t] = (H[rr * N_ + nn] != 0.0f) ? (_Float16)1.0f : (_Float16)0.0f;
}

// ---- per-iteration kernels ------------------------------------------------

__global__ void k_init(const float* __restrict__ llr, float* c2vA,
                       float* totalBuf, int* frozen, float* out) {
    int t = blockIdx.x * blockDim.x + threadIdx.x;
    if (t < B_ * R_ * MAXDEG) c2vA[t] = 0.0f;
    if (t < B_ * N_) {
        totalBuf[t]        = 0.0f;
        out[t]             = llr[t];   // out_llr0 = llr
        out[B_ * N_ + t]   = 0.0f;     // out_bits0 = 0
    }
    if (t < B_) frozen[t] = 0;
}

__global__ void k_clear(int* bad) {
    if (threadIdx.x < B_) bad[threadIdx.x] = 0;
}

// check-node update: C2V_new from llr, total(old C2V), old C2V. Two passes to
// avoid register-indexed local arrays (no scratch spills).
__global__ void k_vc(const float* __restrict__ llr,
                     const float* __restrict__ totalOld,
                     const float* __restrict__ c2vOld, float* __restrict__ c2vNew,
                     const int* __restrict__ rowDeg, const int* __restrict__ rowCols) {
    int t = blockIdx.x * blockDim.x + threadIdx.x;
    if (t >= B_ * R_) return;
    int b = t >> 9;
    int r = t & (R_ - 1);
    int deg = rowDeg[r];
    const int* cols = rowCols + r * MAXDEG;
    long ebase = ((long)b * R_ + r) * MAXDEG;

    float signprod = 1.0f, sumphi = 0.0f;
    for (int s = 0; s < deg; ++s) {
        int c = cols[s];
        float v2c = clipf(llr[b * N_ + c] + GAMMA_ * (totalOld[b * N_ + c] - c2vOld[ebase + s]));
        if (v2c < 0.0f) signprod = -signprod;
        sumphi += phif(fmaxf(fabsf(v2c), 1e-6f));
    }
    for (int s = 0; s < deg; ++s) {
        int c = cols[s];
        float v2c = clipf(llr[b * N_ + c] + GAMMA_ * (totalOld[b * N_ + c] - c2vOld[ebase + s]));
        float phi = phif(fmaxf(fabsf(v2c), 1e-6f));
        float excphi = fmaxf(sumphi - phi, 1e-6f);
        float sgn = (v2c < 0.0f) ? -1.0f : 1.0f;
        c2vNew[ebase + s] = clipf((signprod * sgn) * phif(excphi));
    }
}

// variable-node: total_new + posterior
__global__ void k_var(const float* __restrict__ llr, const float* __restrict__ c2vNew,
                      const int* __restrict__ colDeg, const int* __restrict__ colEdges,
                      float* totalBuf, float* post) {
    int t = blockIdx.x * blockDim.x + threadIdx.x;
    if (t >= B_ * N_) return;
    int b = t >> 10;
    int n = t & (N_ - 1);
    const float* c2v = c2vNew + (long)b * R_ * MAXDEG;
    int cd = colDeg[n];
    const int* es = colEdges + n * MAXDEG;
    float s = 0.0f;
    for (int k = 0; k < cd; ++k) s += c2v[es[k]];
    totalBuf[t] = s;
    post[t] = llr[t] + s;
}

// per-sample mean-|llr| (confidence)
__global__ void k_stats(const float* __restrict__ post, float* sumabs) {
    __shared__ float sm[256];
    int b = blockIdx.x;
    float s = 0.0f;
    for (int i = threadIdx.x; i < N_; i += 256) s += fabsf(post[b * N_ + i]);
    sm[threadIdx.x] = s;
    __syncthreads();
    for (int o = 128; o > 0; o >>= 1) {
        if (threadIdx.x < o) sm[threadIdx.x] += sm[threadIdx.x + o];
        __syncthreads();
    }
    if (threadIdx.x == 0) sumabs[b] = sm[0];
}

// pack hard-decision bits into WMMA A layout (16-bit A 16x32, ISA 7.12.2):
// lanes 0-15: M=lane, K = e + (e>=8 ? 8 : 0); lanes 16-31: M=lane-16, K += 8
__global__ void k_packA(const float* __restrict__ post, _Float16* Apack) {
    int t = blockIdx.x * blockDim.x + threadIdx.x;
    if (t >= KT_ * 32) return;
    int kt = t >> 5;
    int L  = t & 31;
    int m  = L & 15;
    int k0 = kt * 32 + ((L < 16) ? 0 : 8);
    _Float16* ap = Apack + t * 16;
#pragma unroll
    for (int e = 0; e < 16; ++e) {
        int k = k0 + e + ((e >= 8) ? 8 : 0);
        ap[e] = (post[m * N_ + k] < 0.0f) ? (_Float16)1.0f : (_Float16)0.0f;
    }
}

// syndrome = bits(16xN) @ H^T(NxR) via v_wmma_f32_16x16x32_f16, parity test
__global__ void k_synd(const _Float16* __restrict__ Apack,
                       const _Float16* __restrict__ Bpack, int* bad) {
    int j = blockIdx.x;      // check tile: columns jr = j*16 .. +15
    int L = threadIdx.x;     // lane 0..31
    v8f acc = {0.f, 0.f, 0.f, 0.f, 0.f, 0.f, 0.f, 0.f};
    for (int kt = 0; kt < KT_; ++kt) {
        v16h a  = *(const v16h*)(Apack + (kt * 32 + L) * 16);
        v16h bm = *(const v16h*)(Bpack + ((kt * JT_ + j) * 32 + L) * 16);
        acc = __builtin_amdgcn_wmma_f32_16x16x32_f16(
            /*neg_a=*/false, a, /*neg_b=*/false, bm,
            /*c_mod=*/(short)0, acc, /*reuse_a=*/false, /*reuse_b=*/false);
    }
    int mbase = (L >= 16) ? 8 : 0;     // C layout: VGPR i -> M=i (+8 for hi lanes)
#pragma unroll
    for (int i = 0; i < 8; ++i) {
        int v = __float2int_rn(acc[i]);           // exact small integer
        if (v & 1) atomicOr(&bad[mbase + i], 1);  // odd syndrome -> check fails
    }
}

__global__ void k_decide(const int* bad, const float* sumabs, int* frozen, int* newly) {
    int b = threadIdx.x;
    if (b < B_) {
        bool check_ok = (bad[b] == 0);
        bool conf_ok  = (sumabs[b] * (1.0f / N_)) >= CONF_;
        bool nw = check_ok && conf_ok && (frozen[b] == 0);
        newly[b] = nw ? 1 : 0;
        if (nw) frozen[b] = 1;
    }
}

__global__ void k_copy(const float* __restrict__ post, const int* newly, float* out) {
    int t = blockIdx.x * blockDim.x + threadIdx.x;
    if (t >= B_ * N_) return;
    int b = t >> 10;
    if (newly[b]) {
        out[t]           = post[t];
        out[B_ * N_ + t] = (post[t] < 0.0f) ? 1.0f : 0.0f;
    }
}

__global__ void k_final(const float* __restrict__ post, const int* frozen, float* out) {
    int t = blockIdx.x * blockDim.x + threadIdx.x;
    if (t < B_ * N_) {
        int b = t >> 10;
        if (!frozen[b]) {
            out[t]           = post[t];
            out[B_ * N_ + t] = (post[t] < 0.0f) ? 1.0f : 0.0f;
        }
    }
    if (t < B_) out[2 * B_ * N_ + t] = frozen[t] ? 1.0f : 0.0f;
}

// ---------------------------------------------------------------------------

extern "C" void kernel_launch(void* const* d_in, const int* in_sizes, int n_in,
                              void* d_out, int out_size, void* d_ws, size_t ws_size,
                              hipStream_t stream) {
    (void)in_sizes; (void)n_in; (void)out_size; (void)ws_size;
    const float* llr = (const float*)d_in[0];   // [16,1024]
    const float* H   = (const float*)d_in[1];   // [512,1024]
    float* out = (float*)d_out;                 // out_llr | out_bits | frozen

    char* p = (char*)d_ws;
    auto carve = [&](size_t bytes) -> char* {
        char* q = p;
        p += (bytes + 255) & ~(size_t)255;
        return q;
    };
    int*      rowDeg   = (int*)carve(R_ * 4);
    int*      rowCols  = (int*)carve(R_ * MAXDEG * 4);
    int*      colDeg   = (int*)carve(N_ * 4);
    int*      colEdges = (int*)carve(N_ * MAXDEG * 4);
    int*      frozen   = (int*)carve(64);
    int*      bad      = (int*)carve(64);
    int*      newly    = (int*)carve(64);
    float*    sumabs   = (float*)carve(64);
    float*    c2vA     = (float*)carve((size_t)B_ * R_ * MAXDEG * 4);
    float*    c2vB     = (float*)carve((size_t)B_ * R_ * MAXDEG * 4);
    float*    totalBuf = (float*)carve((size_t)B_ * N_ * 4);
    float*    post     = (float*)carve((size_t)B_ * N_ * 4);
    _Float16* Bp       = (_Float16*)carve((size_t)KT_ * JT_ * 32 * 16 * 2);
    _Float16* Ap       = (_Float16*)carve((size_t)KT_ * 32 * 16 * 2);

    // structure + packing (deterministic, rebuilt every call)
    k_rows <<<2,    256, 0, stream>>>(H, rowDeg, rowCols);
    k_cols <<<4,    256, 0, stream>>>(H, rowDeg, rowCols, colDeg, colEdges);
    k_packB<<<2048, 256, 0, stream>>>(H, Bp);
    k_init <<<1024, 256, 0, stream>>>(llr, c2vA, totalBuf, frozen, out);

    // iteration 0: hard check directly on channel LLRs
    k_clear <<<1,  32,  0, stream>>>(bad);
    k_stats <<<16, 256, 0, stream>>>(llr, sumabs);
    k_packA <<<4,  256, 0, stream>>>(llr, Ap);
    k_synd  <<<32, 32,  0, stream>>>(Ap, Bp, bad);
    k_decide<<<1,  32,  0, stream>>>(bad, sumabs, frozen, newly);
    k_copy  <<<64, 256, 0, stream>>>(llr, newly, out);

    float* cOld = c2vA;
    float* cNew = c2vB;
    for (int it = 0; it < NITER; ++it) {
        k_clear <<<1,  32,  0, stream>>>(bad);
        k_vc    <<<32, 256, 0, stream>>>(llr, totalBuf, cOld, cNew, rowDeg, rowCols);
        k_var   <<<64, 256, 0, stream>>>(llr, cNew, colDeg, colEdges, totalBuf, post);
        k_stats <<<16, 256, 0, stream>>>(post, sumabs);
        k_packA <<<4,  256, 0, stream>>>(post, Ap);
        k_synd  <<<32, 32,  0, stream>>>(Ap, Bp, bad);
        k_decide<<<1,  32,  0, stream>>>(bad, sumabs, frozen, newly);
        k_copy  <<<64, 256, 0, stream>>>(post, newly, out);
        float* tmp = cOld; cOld = cNew; cNew = tmp;
    }
    k_final<<<64, 256, 0, stream>>>(post, frozen, out);
}
